// NodeAttentionWithEdgeBias_37048387895505
// MI455X (gfx1250) — compile-verified, matrix operand order checked by reading, not run on hardware
//
#include <hip/hip_runtime.h>
#include <hip/hip_bf16.h>

typedef __bf16 bf16;
typedef bf16  v16bf  __attribute__((ext_vector_type(16)));
typedef bf16  bf16x8 __attribute__((ext_vector_type(8)));
typedef float v8f    __attribute__((ext_vector_type(8)));

#define HEADS 8
#define HD    32
#define NN    768
#define LN_EPS 1e-5f

// ---------------- WMMA helpers ----------------

__device__ __forceinline__ v8f wmma_bf16(v16bf a, v16bf b, v8f c) {
  // D = A(16x32) * B(32x16) + C, f32 accum
  return __builtin_amdgcn_wmma_f32_16x16x32_bf16(false, a, false, b, (short)0, c,
                                                 false, false);
}

__device__ __forceinline__ v8f zero8() {
  v8f z;
#pragma unroll
  for (int i = 0; i < 8; i++) z[i] = 0.f;
  return z;
}

// Load a 16x32 bf16 operand tile from a row-major matrix (row stride in elems).
// Lane L (0..15): row = L,     K = {0..7, 16..23}
// Lane L (16..31): row = L-16, K = {8..15, 24..31}
// Matches the ISA A-layout; B = (other matrix)^T uses the identical pattern.
__device__ __forceinline__ v16bf load_tile(const bf16* base, int stride) {
  int lane = threadIdx.x & 31;
  const bf16* p = base + (lane & 15) * stride + ((lane >> 4) << 3);
  bf16x8 lo = *(const bf16x8*)p;         // K = koff .. koff+7
  bf16x8 hi = *(const bf16x8*)(p + 16);  // K = koff+16 .. koff+23
  return __builtin_shufflevector(lo, hi, 0, 1, 2, 3, 4, 5, 6, 7,
                                 8, 9, 10, 11, 12, 13, 14, 15);
}

// Load one lane's 16-element slice of an f32 A-tile (layout above), converting
// to bf16 while accumulating sum / sum-of-squares of the f32 values.
__device__ __forceinline__ v16bf load_a_f32(const float* p, float& s, float& ss) {
  float4 a = *(const float4*)(p);
  float4 b = *(const float4*)(p + 4);
  float4 c = *(const float4*)(p + 16);
  float4 d = *(const float4*)(p + 20);
  float x[16] = {a.x, a.y, a.z, a.w, b.x, b.y, b.z, b.w,
                 c.x, c.y, c.z, c.w, d.x, d.y, d.z, d.w};
  v16bf r;
#pragma unroll
  for (int t = 0; t < 16; t++) {
    s += x[t];
    ss += x[t] * x[t];
    r[t] = (bf16)x[t];
  }
  return r;
}

__device__ __forceinline__ float wave_red(float v) {
#pragma unroll
  for (int m = 16; m >= 1; m >>= 1) v += __shfl_xor(v, m, 32);
  return v;
}

// ---------------- Phase 0a: pack weights to bf16, transposed ----------------
// Wt[c][k] = W[k][c]  (so WMMA B-operand loads are contiguous b128)
__global__ void __launch_bounds__(256)
pack_weights(const float* wq, const float* wk, const float* wv,
             const float* wg, const float* wo,
             bf16* tq, bf16* tk, bf16* tv, bf16* tg, bf16* to_) {
  int idx = blockIdx.x * blockDim.x + threadIdx.x;   // 5 * 65536 total
  int m = idx >> 16;
  int o = idx & 65535;
  int c = o >> 8, kk = o & 255;
  const float* s; bf16* d;
  switch (m) {
    case 0: s = wq; d = tq; break;
    case 1: s = wk; d = tk; break;
    case 2: s = wv; d = tv; break;
    case 3: s = wg; d = tg; break;
    default: s = wo; d = to_; break;
  }
  d[o] = (bf16)s[kk * 256 + c];
}

// ---------------- Phase 0b: edge-projection prep ----------------
// wwbt[h][e] = lnw[e] * Wb[e][h]   (h in [0,16), zero-padded h>=8), bf16
// cwb[h] = sum_e lnw[e]*Wb[e][h] ; cbb[h] = sum_e lnb[e]*Wb[e][h]
__global__ void __launch_bounds__(256)
prep_edge(const float* lnw, const float* lnb, const float* Wb,
          bf16* wwbt, float* cwb, float* cbb) {
  int tid = threadIdx.x;
  for (int idx = tid; idx < 2048; idx += 256) {
    int h = idx >> 7, e = idx & 127;
    wwbt[idx] = (bf16)((h < 8) ? lnw[e] * Wb[e * 8 + h] : 0.f);
  }
  if (tid < 16) {
    float s = 0.f;
    if (tid < 8)
      for (int e = 0; e < 128; e++) s += lnw[e] * Wb[e * 8 + tid];
    cwb[tid] = s;
  } else if (tid < 32) {
    int h = tid - 16;
    float s = 0.f;
    if (h < 8)
      for (int e = 0; e < 128; e++) s += lnb[e] * Wb[e * 8 + h];
    cbb[h] = s;
  }
}

// ---------------- Phase 1: node LN + Q/K/V/G projections ----------------
__global__ void __launch_bounds__(128)
ln_qkvg(const float* node, const float* lnw, const float* lnb,
        const bf16* tq, const bf16* tk, const bf16* tv, const bf16* tg,
        const float* bg,
        bf16* q, bf16* k, bf16* v_t, float* g) {
  __shared__ bf16 xln[16 * 256];
  int tid = threadIdx.x;
  int row0 = blockIdx.x * 16;   // global row in [0, 1536)

  { // LayerNorm: 8 lanes per row, 32 elems per lane
    int r = tid >> 3, sub = tid & 7;
    const float* src = node + (size_t)(row0 + r) * 256 + sub * 32;
    float x[32];
#pragma unroll
    for (int t = 0; t < 32; t += 4) {
      float4 f = *(const float4*)(src + t);
      x[t] = f.x; x[t + 1] = f.y; x[t + 2] = f.z; x[t + 3] = f.w;
    }
    float s = 0.f, ss = 0.f;
#pragma unroll
    for (int t = 0; t < 32; t++) { s += x[t]; ss += x[t] * x[t]; }
#pragma unroll
    for (int m = 4; m >= 1; m >>= 1) {
      s  += __shfl_xor(s, m, 8);
      ss += __shfl_xor(ss, m, 8);
    }
    float mu = s * (1.f / 256.f);
    float rs = rsqrtf(ss * (1.f / 256.f) - mu * mu + LN_EPS);
#pragma unroll
    for (int t = 0; t < 32; t++) {
      int c = sub * 32 + t;
      xln[r * 256 + c] = (bf16)((x[t] - mu) * rs * lnw[c] + lnb[c]);
    }
  }
  __syncthreads();

  int lane = tid & 31, w = tid >> 5;
  int j_l = lane & 15, i_base = (lane >> 4) * 8;
  for (int nt = w; nt < 64; nt += 4) {          // 64 n-tiles over [Wq|Wk|Wv|Wg]
    int mat = nt >> 4;                          // uniform per wave iteration
    const bf16* wt = (mat == 0) ? tq : (mat == 1) ? tk : (mat == 2) ? tv : tg;
    int colbase = (nt & 15) * 16;
    v8f acc = zero8();
#pragma unroll
    for (int kc = 0; kc < 8; kc++) {
      v16bf A  = load_tile(&xln[kc * 32], 256);
      v16bf Bm = load_tile(wt + colbase * 256 + kc * 32, 256);
      acc = wmma_bf16(A, Bm, acc);
    }
    int col = colbase + j_l;                    // column within matrix [0,256)
    int h = col >> 5, d = col & 31;
#pragma unroll
    for (int r = 0; r < 8; r++) {
      int ig = row0 + i_base + r;               // global row
      int b = ig / NN, i = ig % NN;
      float val = acc[r];
      if (mat == 0)      q  [(((b * HEADS + h) * NN + i) << 5) + d] = (bf16)val;
      else if (mat == 1) k  [(((b * HEADS + h) * NN + i) << 5) + d] = (bf16)val;
      else if (mat == 2) v_t[(((b * HEADS + h) << 5) + d) * NN + i] = (bf16)val;
      else               g  [(((b * HEADS + h) * NN + i) << 5) + d] =
                             1.f / (1.f + __expf(-(val + bg[col])));
    }
  }
}

// ---------------- Phase 2: edge LN + bias projection (WMMA streaming) -----
// bias[b][h][i][j] = LN(edge[b,i,j,:]) @ Wb, LN folded as a rank-1 fixup:
//   T = y_raw @ (lnw .* Wb)   (4 WMMAs over a 16-j tile)
//   bias = rs_row * (T - mu_row * cw) + cb
__global__ void __launch_bounds__(256)
edge_bias(const float* edge, const bf16* wwbt, const float* cwb,
          const float* cbb, float* bias) {
  __shared__ float buf[8][8][16];               // [wave][h][jj]
  int tid = threadIdx.x, lane = tid & 31, w = tid >> 5;
  int chunk = blockIdx.x * 8 + w;               // 73728 chunks of 16 rows, exact
  size_t row0 = (size_t)chunk * 16;
  int b  = (int)(row0 / ((size_t)NN * NN));
  int i  = (int)((row0 / NN) % NN);
  int j0 = (int)(row0 % NN);                    // multiple of 16

  // B operand: (lnw .* Wb), 128x16 as four 32x16 tiles (L2-resident, 4 KB)
  v16bf B0 = load_tile(wwbt + 0,  128);
  v16bf B1 = load_tile(wwbt + 32, 128);
  v16bf B2 = load_tile(wwbt + 64, 128);
  v16bf B3 = load_tile(wwbt + 96, 128);
  float cw_l = cwb[lane & 15];
  float cb_l = cbb[lane & 15];

  // Each lane streams its row's K-subset; lane L & L+16 are complementary.
  const float* yp = edge + (row0 + (lane & 15)) * 128 + ((lane >> 4) << 3);
  float s = 0.f, ss = 0.f;
  v16bf A0 = load_a_f32(yp + 0,  s, ss);
  v16bf A1 = load_a_f32(yp + 32, s, ss);
  v16bf A2 = load_a_f32(yp + 64, s, ss);
  v16bf A3 = load_a_f32(yp + 96, s, ss);
  s  += __shfl_xor(s, 16, 32);
  ss += __shfl_xor(ss, 16, 32);
  float mu = s * (1.f / 128.f);
  float rs = rsqrtf(ss * (1.f / 128.f) - mu * mu + LN_EPS);

  v8f acc = zero8();
  acc = wmma_bf16(A0, B0, acc);
  acc = wmma_bf16(A1, B1, acc);
  acc = wmma_bf16(A2, B2, acc);
  acc = wmma_bf16(A3, B3, acc);

  int i_base = (lane >> 4) * 8;                 // D-layout row base
  int hsel = lane & 15;                         // D-layout column = head
#pragma unroll
  for (int r = 0; r < 8; r++) {
    float mu_r = __shfl(mu, i_base + r, 32);    // row stats live in lane = row
    float rs_r = __shfl(rs, i_base + r, 32);
    float val = rs_r * (acc[r] - mu_r * cw_l) + cb_l;
    if (hsel < 8) buf[w][hsel][i_base + r] = val;
  }
  __syncthreads();
  if (lane < 16) {
#pragma unroll
    for (int h = 0; h < 8; h++)
      bias[((size_t)(b * HEADS + h) * NN + i) * NN + j0 + lane] = buf[w][h][lane];
  }
}

// ---------------- Phase 3: attention per (b, h, 16-row i-tile) ------------
__global__ void __launch_bounds__(128)
attention(const bf16* q, const bf16* k, const bf16* v_t, const float* g,
          const float* bias, bf16* res) {
  __shared__ float logits[16 * 768];
  __shared__ float rowsum[16];
  __shared__ float part[16][32][4];
  int tid = threadIdx.x, lane = tid & 31, w = tid >> 5;
  int it = blockIdx.x % 48;
  int h  = (blockIdx.x / 48) & 7;
  int b  = blockIdx.x / 384;
  int bh = b * HEADS + h;
  const float scale = 0.17677669529663687f;     // 1/sqrt(32)
  int j_l = lane & 15, i_base = (lane >> 4) * 8;

  v16bf Aq = load_tile(q + ((size_t)(bh * NN + it * 16) << 5), 32);

  // Pass 1: logits strip (16 x 768) in LDS. One WMMA per j-tile (K = HD = 32).
  for (int jt = w; jt < 48; jt += 4) {
    v16bf Bk = load_tile(k + ((size_t)(bh * NN + jt * 16) << 5), 32);
    v8f acc = zero8();
    acc = wmma_bf16(Aq, Bk, acc);
    const float* bp = bias + ((size_t)bh * NN + it * 16) * NN + jt * 16 + j_l;
    __builtin_prefetch(bp + 64, 0, 0);          // next jt strip -> global_prefetch
#pragma unroll
    for (int r = 0; r < 8; r++) {
      int i_l = i_base + r;
      logits[i_l * NN + jt * 16 + j_l] = acc[r] * scale + bp[(size_t)i_l * NN];
    }
  }
  __syncthreads();

  // Stable softmax over j: wave w owns rows {4w .. 4w+3}
#pragma unroll
  for (int rr = 0; rr < 4; rr++) {
    int row = w * 4 + rr;
    float m = -3.0e38f;
    for (int t = lane; t < NN; t += 32) m = fmaxf(m, logits[row * NN + t]);
#pragma unroll
    for (int mm = 16; mm >= 1; mm >>= 1) m = fmaxf(m, __shfl_xor(m, mm, 32));
    float s = 0.f;
    for (int t = lane; t < NN; t += 32) {
      float p = __expf(logits[row * NN + t] - m);
      logits[row * NN + t] = p;
      s += p;
    }
    s = wave_red(s);
    if (lane == 0) rowsum[row] = s;
  }
  __syncthreads();

  // Pass 2: out = P @ V ; P re-read from LDS as bf16 A-operand, V^T contiguous.
  v8f acc0 = zero8(), acc1 = zero8();
  const bf16* vb = v_t + ((size_t)bh << 5) * NN;
  int prow = lane & 15, koff = (lane >> 4) * 8;
  for (int kc = w; kc < 24; kc += 4) {
    int j0 = kc * 32;
    const float* lp = &logits[prow * NN + j0 + koff];
    v16bf A;
#pragma unroll
    for (int t = 0; t < 8; t++) A[t] = (bf16)lp[t];
#pragma unroll
    for (int t = 0; t < 8; t++) A[8 + t] = (bf16)lp[16 + t];
    v16bf B0 = load_tile(vb + j0, NN);             // d in [0,16)
    v16bf B1 = load_tile(vb + 16 * NN + j0, NN);   // d in [16,32)
    acc0 = wmma_bf16(A, B0, acc0);
    acc1 = wmma_bf16(A, B1, acc1);
  }
#pragma unroll
  for (int r = 0; r < 8; r++) {
    part[i_base + r][j_l][w]      = acc0[r];
    part[i_base + r][16 + j_l][w] = acc1[r];
  }
  __syncthreads();

  // Reduce 4-wave partials, apply gate and 1/l, store res[b][i][h*32+d] (bf16)
#pragma unroll
  for (int t = 0; t < 4; t++) {
    int e = tid * 4 + t;
    int i_l = e >> 5, d = e & 31;
    float s = part[i_l][d][0] + part[i_l][d][1] + part[i_l][d][2] + part[i_l][d][3];
    int ig = it * 16 + i_l;
    float gate = g[(((size_t)(bh * NN + ig)) << 5) + d];
    float val = s * gate / rowsum[i_l];
    res[((size_t)(b * NN + ig) << 8) + h * 32 + d] = (bf16)val;
  }
}

// ---------------- Phase 4: output projection res @ Wo + bo ----------------
__global__ void __launch_bounds__(128)
out_proj(const bf16* res, const bf16* wo_t, const float* bo, float* out) {
  int tid = threadIdx.x, lane = tid & 31, w = tid >> 5;
  int row0 = blockIdx.x * 16;
  int j_l = lane & 15, i_base = (lane >> 4) * 8;
  for (int nt = w; nt < 16; nt += 4) {
    v8f acc = zero8();
#pragma unroll
    for (int kc = 0; kc < 8; kc++) {
      v16bf A  = load_tile(res + (size_t)row0 * 256 + kc * 32, 256);
      v16bf Bm = load_tile(wo_t + nt * 16 * 256 + kc * 32, 256);
      acc = wmma_bf16(A, Bm, acc);
    }
    int c = nt * 16 + j_l;
    float bov = bo[c];
#pragma unroll
    for (int r = 0; r < 8; r++)
      out[(size_t)(row0 + i_base + r) * 256 + c] = acc[r] + bov;
  }
}

// ---------------- launch ----------------

extern "C" void kernel_launch(void* const* d_in, const int* in_sizes, int n_in,
                              void* d_out, int out_size, void* d_ws, size_t ws_size,
                              hipStream_t stream) {
  const float* node      = (const float*)d_in[0];
  const float* edge      = (const float*)d_in[1];
  const float* ln_node_w = (const float*)d_in[2];
  const float* ln_node_b = (const float*)d_in[3];
  const float* ln_edge_w = (const float*)d_in[4];
  const float* ln_edge_b = (const float*)d_in[5];
  const float* Wq        = (const float*)d_in[6];
  const float* Wk        = (const float*)d_in[7];
  const float* Wv        = (const float*)d_in[8];
  const float* Wb        = (const float*)d_in[9];
  const float* Wg        = (const float*)d_in[10];
  const float* bg        = (const float*)d_in[11];
  const float* Wo        = (const float*)d_in[12];
  const float* bo        = (const float*)d_in[13];
  float* out = (float*)d_out;

  char* ws = (char*)d_ws;
  bf16*  tq   = (bf16*)(ws + 0);          // 256x256 bf16 each (transposed)
  bf16*  tk   = (bf16*)(ws + 131072);
  bf16*  tv   = (bf16*)(ws + 262144);
  bf16*  tg   = (bf16*)(ws + 393216);
  bf16*  to_  = (bf16*)(ws + 524288);
  bf16*  qb   = (bf16*)(ws + 655360);     // [b][h][i][d] bf16
  bf16*  kb   = (bf16*)(ws + 1441792);    // [b][h][j][d] bf16
  bf16*  vtb  = (bf16*)(ws + 2228224);    // [b][h][d][j] bf16
  float* gb   = (float*)(ws + 3014656);   // [b][h][i][d] f32
  float* bb   = (float*)(ws + 4587520);   // [b][h][i][j] f32 (36 MB)
  bf16*  resb = (bf16*)(ws + 42336256);   // [b][i][hidden] bf16
  bf16*  wwbt = (bf16*)(ws + 43122688);   // 16x128 bf16 (lnw .* Wb, padded)
  float* cwb  = (float*)(ws + 43126784);  // 16 f32
  float* cbb  = (float*)(ws + 43126848);  // 16 f32

  pack_weights<<<1280, 256, 0, stream>>>(Wq, Wk, Wv, Wg, Wo, tq, tk, tv, tg, to_);
  prep_edge<<<1, 256, 0, stream>>>(ln_edge_w, ln_edge_b, Wb, wwbt, cwb, cbb);
  ln_qkvg<<<96, 128, 0, stream>>>(node, ln_node_w, ln_node_b,
                                  tq, tk, tv, tg, bg, qb, kb, vtb, gb);
  edge_bias<<<9216, 256, 0, stream>>>(edge, wwbt, cwb, cbb, bb);
  attention<<<768, 128, 0, stream>>>(qb, kb, vtb, gb, bb, resb);
  out_proj<<<96, 128, 0, stream>>>(resb, to_, bo, out);
}
